// SecondOrderDeformableAlignment_21449066676749
// MI455X (gfx1250) — matrix operand order repaired and checked by reading, not compiled, not run on hardware
//
#include <hip/hip_runtime.h>
#include <math.h>

typedef __bf16 bf16;
typedef __attribute__((ext_vector_type(16))) __bf16 v16bf;
typedef __attribute__((ext_vector_type(4)))  __bf16 v4bf;
typedef __attribute__((ext_vector_type(8)))  float  v8f;

// Problem geometry (fixed by the reference harness)
#define BB   2
#define HHT  96
#define WWT  160
#define W16  (WWT/16)
#define NPIX (HHT*WWT)
#define NBLK (BB*HHT*W16)   // 1920 pixel tiles of 16 (one row segment each)
#define PR   24             // padded patch row stride (elements); col c -> offset c+3

// ---------------------------------------------------------------------------
// Repack fp32 conv weights (Cout,Cin,3,3) into pre-swizzled bf16 A-fragments.
// Fragment (mt,kc) = 512 bf16, element index = lane*16+e.
// A-layout (16-bit, this code's contiguous k order, matched by B): m = lane&15 ; k_off = e + 16*(lane>>4)
// K mapping: k = ci*9 + tap, zero-padded past Cin*9.
// ---------------------------------------------------------------------------
__global__ __launch_bounds__(256)
void repack_w(const float* __restrict__ W, bf16* __restrict__ out,
              int Cout, int Cin, int MT, int KC)
{
    int t = blockIdx.x * 256 + threadIdx.x;
    int total = MT * KC * 512;
    if (t >= total) return;
    int frag = t >> 9, r = t & 511;
    int lane = r >> 4, e = r & 15;
    int mt = frag / KC, kc = frag - mt * KC;
    int m = mt * 16 + (lane & 15);
    int k = kc * 32 + e + 16 * (lane >> 4);
    int Ktot = Cin * 9;
    float v = 0.f;
    if (m < Cout && k < Ktot) {
        int ci = k / 9, tap = k - ci * 9;
        v = W[((size_t)m * Cin + ci) * 9 + tap];
    }
    out[t] = (bf16)v;
}

// ---------------------------------------------------------------------------
// Implicit-GEMM 3x3 conv (stride1 pad1) via WMMA bf16.
//   M = Cout (16/wave, 8 waves/block, grid.y tiles of 8)
//   N = 16 horizontally-consecutive pixels, K = Cin*9 in chunks of 64 (2 WMMA).
// The block's full Cin x 3 x 18 halo patch is preloaded (coalesced) into LDS
// as bf16; the K-loop im2col staging is then pure LDS traffic.
// MODE 0: input = virtual concat [extra_feat(384)|flow1(2)|flow2(2)] fp32
// MODE 1: input = bf16 activations, epilogue lrelu -> bf16
// MODE 2: input = bf16 activations, epilogue offset(tanh+flow)/sigmoid mask
// ---------------------------------------------------------------------------
template<int MODE>
__global__ __launch_bounds__(256)
void conv_wmma(const bf16* __restrict__ inB, const float* __restrict__ extra,
               const float* __restrict__ flow1, const float* __restrict__ flow2,
               const bf16* __restrict__ pw, const float* __restrict__ bias,
               bf16* __restrict__ outB, float* __restrict__ outOff,
               float* __restrict__ outMask,
               int Cout, int KC, int Ktot)
{
    constexpr int CIN = (MODE == 0) ? 388 : 128;
    __shared__ __align__(32) bf16 Bt[1024];          // two 32x16 B fragments
    __shared__ __align__(16) bf16 patch[CIN * 3 * PR];

    const int pt  = blockIdx.x;
    const int b   = pt / (HHT * W16);
    const int rm  = pt - b * (HHT * W16);
    const int y   = rm / W16;
    const int x0  = (rm - y * W16) * 16;
    const int tid = threadIdx.x;
    const int lane = tid & 31;
    const int wv   = tid >> 5;
    const int mt   = blockIdx.y * 8 + wv;
    const int MT   = (Cout + 15) >> 4;
    const bool active = (mt < MT);

    // ---- preload halo patch: interior cols (coalesced 4-wide) ----
    for (int i = tid; i < CIN * 3 * 4; i += 256) {
        int q = i & 3, cr = i >> 2;
        int row = cr % 3, ci = cr / 3;
        int yy = y + row - 1;
        v4bf val;
        if (yy >= 0 && yy < HHT) {
            int xx = x0 + q * 4;
            if (MODE == 0) {
                const float* src; int cc;
                if (ci < 384)      { src = extra; cc = b * 384 + ci; }
                else if (ci < 386) { src = flow1; cc = b * 2 + (ci - 384); }
                else               { src = flow2; cc = b * 2 + (ci - 386); }
                float4 f = *(const float4*)&src[((size_t)cc * HHT + yy) * WWT + xx];
                val[0] = (bf16)f.x; val[1] = (bf16)f.y;
                val[2] = (bf16)f.z; val[3] = (bf16)f.w;
            } else {
                val = *(const v4bf*)&inB[((size_t)(b * CIN + ci) * HHT + yy) * WWT + xx];
            }
        } else {
            val[0] = (bf16)0.f; val[1] = (bf16)0.f;
            val[2] = (bf16)0.f; val[3] = (bf16)0.f;
        }
        *(v4bf*)&patch[(ci * 3 + row) * PR + 4 + q * 4] = val;
    }
    // ---- preload halo patch: border cols (c=0 -> off 3, c=17 -> off 20) ----
    for (int i = tid; i < CIN * 3 * 2; i += 256) {
        int side = i & 1, cr = i >> 1;
        int row = cr % 3, ci = cr / 3;
        int yy = y + row - 1;
        int xx = side ? (x0 + 16) : (x0 - 1);
        float v = 0.f;
        if (yy >= 0 && yy < HHT && xx >= 0 && xx < WWT) {
            if (MODE == 0) {
                if (ci < 384)      v = extra[((size_t)(b*384 + ci)*HHT + yy)*WWT + xx];
                else if (ci < 386) v = flow1[((size_t)(b*2 + (ci-384))*HHT + yy)*WWT + xx];
                else               v = flow2[((size_t)(b*2 + (ci-386))*HHT + yy)*WWT + xx];
            } else {
                v = (float)inB[((size_t)(b * CIN + ci) * HHT + yy) * WWT + xx];
            }
        }
        patch[(ci * 3 + row) * PR + 3 + (side ? 17 : 0)] = (bf16)v;
    }

    v8f acc = {0.f,0.f,0.f,0.f,0.f,0.f,0.f,0.f};
    const int KC2 = KC >> 1;                     // KC is even for all layers

    for (int kc2 = 0; kc2 < KC2; ++kc2) {
        __syncthreads();
        // ---- stage two im2col B fragments (64 K x 16 N) from the patch ----
        #pragma unroll
        for (int it = 0; it < 4; ++it) {
            int idx = tid + it * 256;            // 0..1023
            int kl2 = idx >> 4, n = idx & 15;
            int k = kc2 * 64 + kl2;
            bf16 v = (bf16)0.f;
            if (k < Ktot) {
                int ci = k / 9, tap = k - ci * 9;
                int ky = tap / 3, kx = tap - ky * 3;
                v = patch[(ci * 3 + ky) * PR + 3 + kx + n];
            }
            int j  = kl2 >> 5;                   // which fragment
            int kl = kl2 & 31;
            // B layout: lanes 0-15 col n K=e ; lanes 16-31 col n K=16+e
            Bt[j * 512 + (n + ((kl >> 4) << 4)) * 16 + (kl & 15)] = v;
        }
        __syncthreads();
        if (active) {
            const bf16* pa = pw + ((size_t)mt * KC + kc2 * 2) * 512;
            v16bf a0 = *reinterpret_cast<const v16bf*>(pa + lane * 16);
            v16bf a1 = *reinterpret_cast<const v16bf*>(pa + 512 + lane * 16);
            v16bf b0 = *reinterpret_cast<const v16bf*>(&Bt[lane * 16]);
            v16bf b1 = *reinterpret_cast<const v16bf*>(&Bt[512 + lane * 16]);
            acc = __builtin_amdgcn_wmma_f32_16x16x32_bf16(
                false, a0, false, b0, (short)0, acc, false, false);
            acc = __builtin_amdgcn_wmma_f32_16x16x32_bf16(
                false, a1, false, b1, (short)0, acc, false, false);
        }
    }

    if (!active) return;
    const int n  = lane & 15;
    const int xo = x0 + n;
    const int mh = (lane >> 4) << 3;             // lanes>=16 hold M rows 8..15
    #pragma unroll
    for (int r = 0; r < 8; ++r) {
        const int co = mt * 16 + mh + r;
        float val = acc[r] + bias[co];
        if (MODE == 0 || MODE == 1) {
            float o = val >= 0.f ? val : 0.1f * val;
            outB[((size_t)(b * Cout + co) * HHT + y) * WWT + xo] = (bf16)o;
        } else {
            if (co < 288) {
                float t = 10.f * tanhf(val);
                const float* fl = (co < 144) ? flow1 : flow2;
                int c   = (co < 144) ? co : co - 144;
                int fch = 1 - (c & 1);           // flow[:, ::-1] tiling
                t += fl[((size_t)(b * 2 + fch) * HHT + y) * WWT + xo];
                outOff[((size_t)(b * 288 + co) * HHT + y) * WWT + xo] = t;
            } else {
                outMask[((size_t)(b * 144 + (co - 288)) * HHT + y) * WWT + xo] =
                    1.f / (1.f + expf(-val));
            }
        }
    }
}

// ---------------------------------------------------------------------------
// Modulated deformable conv: gather bilinear samples of x (masked) as the
// im2col B tiles (2 chunks / barrier), GEMM with packed weights via WMMA.
// Branchless bilinear: clamped indices, validity folded into the weights
// (matches reference semantics: sample * valid with clipped gather index).
// Cout = Cin = 128, dg = 16, K = 1152 (18 double-chunks).
// ---------------------------------------------------------------------------
__global__ __launch_bounds__(256)
void deform_wmma(const float* __restrict__ x, const float* __restrict__ off,
                 const float* __restrict__ msk, const bf16* __restrict__ pw,
                 const float* __restrict__ bias, float* __restrict__ out)
{
    __shared__ __align__(32) bf16 Bt[1024];
    __shared__ float sPy[16*9*16];   // [dg][tap][n] absolute sample y
    __shared__ float sPx[16*9*16];
    __shared__ float sMk[16*9*16];

    const int pt  = blockIdx.x;
    const int b   = pt / (HHT * W16);
    const int rm  = pt - b * (HHT * W16);
    const int y   = rm / W16;
    const int x0  = (rm - y * W16) * 16;
    const int tid = threadIdx.x;
    const int lane = tid & 31;
    const int mt   = tid >> 5;       // 8 waves -> 8 M-tiles = 128 channels

    // prefetch per-tile offsets / masks (16 dg x 9 taps x 16 pixels)
    for (int idx = tid; idx < 2304; idx += 256) {
        int dg  = idx / 144;
        int r2  = idx - dg * 144;
        int tap = r2 >> 4;
        int n   = r2 & 15;
        int ch  = dg * 9 + tap;
        size_t base = ((size_t)(b * 288 + ch * 2) * HHT + y) * WWT + x0 + n;
        float dy = off[base];
        float dx = off[base + (size_t)HHT * WWT];
        sPy[idx] = dy + (float)(y + tap / 3 - 1);
        sPx[idx] = dx + (float)(x0 + n + (tap % 3) - 1);
        sMk[idx] = msk[((size_t)(b * 144 + ch) * HHT + y) * WWT + x0 + n];
    }

    v8f acc = {0.f,0.f,0.f,0.f,0.f,0.f,0.f,0.f};

    for (int kc2 = 0; kc2 < 18; ++kc2) {         // K = 128*9 = 1152, 64/iter
        __syncthreads();
        #pragma unroll
        for (int it = 0; it < 4; ++it) {
            int idx = tid + it * 256;
            int kl2 = idx >> 4, n = idx & 15;
            int k = kc2 * 64 + kl2;
            int ci = k / 9, tap = k - ci * 9;
            int dg = ci >> 3;                    // cg = 128/16 = 8
            int oi = (dg * 9 + tap) * 16 + n;
            float py = sPy[oi], px = sPx[oi], m = sMk[oi];
            float y0f = floorf(py), x0f = floorf(px);
            float wy = py - y0f,   wx = px - x0f;
            int yi = (int)y0f, xi = (int)x0f;
            // validity folded into bilinear weights; indices clamped so the
            // 4 gathers are unconditional (clause-able, no exec-mask branches)
            float wy0 = (yi     >= 0 && yi     < HHT) ? (1.f - wy) : 0.f;
            float wy1 = (yi + 1 >= 0 && yi + 1 < HHT) ? wy         : 0.f;
            float wx0 = (xi     >= 0 && xi     < WWT) ? (1.f - wx) : 0.f;
            float wx1 = (xi + 1 >= 0 && xi + 1 < WWT) ? wx         : 0.f;
            int yc0 = min(max(yi,     0), HHT - 1);
            int yc1 = min(max(yi + 1, 0), HHT - 1);
            int xc0 = min(max(xi,     0), WWT - 1);
            int xc1 = min(max(xi + 1, 0), WWT - 1);
            const float* xp = x + ((size_t)b * 128 + ci) * NPIX;
            float v00 = xp[yc0 * WWT + xc0];
            float v01 = xp[yc0 * WWT + xc1];
            float v10 = xp[yc1 * WWT + xc0];
            float v11 = xp[yc1 * WWT + xc1];
            float s = wy0 * (wx0 * v00 + wx1 * v01)
                    + wy1 * (wx0 * v10 + wx1 * v11);
            int j  = kl2 >> 5;
            int kl = kl2 & 31;
            Bt[j * 512 + (n + ((kl >> 4) << 4)) * 16 + (kl & 15)] = (bf16)(m * s);
        }
        __syncthreads();
        const bf16* pa = pw + ((size_t)mt * 36 + kc2 * 2) * 512;
        v16bf a0 = *reinterpret_cast<const v16bf*>(pa + lane * 16);
        v16bf a1 = *reinterpret_cast<const v16bf*>(pa + 512 + lane * 16);
        v16bf b0 = *reinterpret_cast<const v16bf*>(&Bt[lane * 16]);
        v16bf b1 = *reinterpret_cast<const v16bf*>(&Bt[512 + lane * 16]);
        acc = __builtin_amdgcn_wmma_f32_16x16x32_bf16(
            false, a0, false, b0, (short)0, acc, false, false);
        acc = __builtin_amdgcn_wmma_f32_16x16x32_bf16(
            false, a1, false, b1, (short)0, acc, false, false);
    }

    const int n  = lane & 15;
    const int xo = x0 + n;
    const int mh = (lane >> 4) << 3;
    #pragma unroll
    for (int r = 0; r < 8; ++r) {
        const int co = mt * 16 + mh + r;
        out[((size_t)(b * 128 + co) * HHT + y) * WWT + xo] = acc[r] + bias[co];
    }
}

// ---------------------------------------------------------------------------
extern "C" void kernel_launch(void* const* d_in, const int* in_sizes, int n_in,
                              void* d_out, int out_size, void* d_ws, size_t ws_size,
                              hipStream_t stream)
{
    const float* x     = (const float*)d_in[0];
    const float* extra = (const float*)d_in[1];
    const float* fl1   = (const float*)d_in[2];
    const float* fl2   = (const float*)d_in[3];
    const float* wgt   = (const float*)d_in[4];
    const float* bias  = (const float*)d_in[5];
    const float* cw0   = (const float*)d_in[6];
    const float* cb0   = (const float*)d_in[7];
    const float* cw1   = (const float*)d_in[8];
    const float* cb1   = (const float*)d_in[9];
    const float* cw2   = (const float*)d_in[10];
    const float* cb2   = (const float*)d_in[11];
    const float* cw3   = (const float*)d_in[12];
    const float* cb3   = (const float*)d_in[13];

    // workspace carve-up (256B aligned)
    char*  ws = (char*)d_ws;
    size_t o  = 0;
    auto carve = [&](size_t bytes) -> char* {
        o = (o + 255) & ~(size_t)255;
        char* p = ws + o;
        o += bytes;
        return p;
    };
    const int KC0 = (388 * 9 + 31) / 32;   // 110 (even)
    const int KC1 = 36;                    // 128*9/32
    bf16*  pw0  = (bf16*)carve((size_t)8  * KC0 * 512 * 2);
    bf16*  pw1  = (bf16*)carve((size_t)8  * KC1 * 512 * 2);
    bf16*  pw2  = (bf16*)carve((size_t)8  * KC1 * 512 * 2);
    bf16*  pw3  = (bf16*)carve((size_t)27 * KC1 * 512 * 2);
    bf16*  pwD  = (bf16*)carve((size_t)8  * KC1 * 512 * 2);
    bf16*  hA   = (bf16*)carve((size_t)BB * 128 * NPIX * 2);
    bf16*  hB   = (bf16*)carve((size_t)BB * 128 * NPIX * 2);
    float* offB = (float*)carve((size_t)BB * 288 * NPIX * 4);
    float* mskB = (float*)carve((size_t)BB * 144 * NPIX * 4);
    (void)ws_size; (void)in_sizes; (void)n_in; (void)out_size;

    auto rpGrid = [](int MT, int KC) { return dim3((MT * KC * 512 + 255) / 256); };
    repack_w<<<rpGrid(8,  KC0), 256, 0, stream>>>(cw0, pw0, 128, 388, 8,  KC0);
    repack_w<<<rpGrid(8,  KC1), 256, 0, stream>>>(cw1, pw1, 128, 128, 8,  KC1);
    repack_w<<<rpGrid(8,  KC1), 256, 0, stream>>>(cw2, pw2, 128, 128, 8,  KC1);
    repack_w<<<rpGrid(27, KC1), 256, 0, stream>>>(cw3, pw3, 432, 128, 27, KC1);
    repack_w<<<rpGrid(8,  KC1), 256, 0, stream>>>(wgt, pwD, 128, 128, 8,  KC1);

    // conv0: [extra|flow1|flow2] (388ch) -> hA (128ch, lrelu)
    conv_wmma<0><<<dim3(NBLK, 1), 256, 0, stream>>>(
        nullptr, extra, fl1, fl2, pw0, cb0, hA, nullptr, nullptr,
        128, KC0, 388 * 9);
    // conv1: hA -> hB
    conv_wmma<1><<<dim3(NBLK, 1), 256, 0, stream>>>(
        hA, nullptr, nullptr, nullptr, pw1, cb1, hB, nullptr, nullptr,
        128, KC1, 128 * 9);
    // conv2: hB -> hA
    conv_wmma<1><<<dim3(NBLK, 1), 256, 0, stream>>>(
        hB, nullptr, nullptr, nullptr, pw2, cb2, hA, nullptr, nullptr,
        128, KC1, 128 * 9);
    // conv3: hA -> offset (fp32, tanh*10 + reversed flow) + mask (sigmoid)
    conv_wmma<2><<<dim3(NBLK, 4), 256, 0, stream>>>(
        hA, nullptr, fl1, fl2, pw3, cb3, nullptr, offB, mskB,
        432, KC1, 128 * 9);
    // deformable conv -> d_out (fp32)
    deform_wmma<<<dim3(NBLK, 1), 256, 0, stream>>>(
        x, offB, mskB, pwD, bias, (float*)d_out);
}
